// CausalSelfAttention_54881092108936
// MI455X (gfx1250) — matrix operand order repaired and checked by reading, not compiled, not run on hardware
//
#include <hip/hip_runtime.h>

// Problem constants (CausalSelfAttention): B=4, S=2048, D=1024, H=16, HD=64
#define BATCH  4
#define SEQ    2048
#define DMODEL 1024
#define NHEADS 16
#define HDIM   64
#define MROWS  (BATCH * SEQ)   // 8192
#define NEGINF (-3.0e38f)
// 1/sqrt(HD) * log2(e): fold softmax base-2 conversion into the Q projection
#define QSCALE (0.125f * 1.4426950408889634f)

typedef __attribute__((ext_vector_type(8)))  __bf16 bf16x8;
typedef __attribute__((ext_vector_type(16))) __bf16 bf16x16;
typedef __attribute__((ext_vector_type(8)))  float  f32x8;

static __device__ __forceinline__ bf16x16 bmerge(bf16x8 lo, bf16x8 hi) {
  bf16x16 r;
#pragma unroll
  for (int i = 0; i < 8; ++i) { r[i] = lo[i]; r[i + 8] = hi[i]; }
  return r;
}

static __device__ __forceinline__ f32x8 wmma_bf16(bf16x16 a, bf16x16 b, f32x8 c) {
  // D = A(16x32 bf16) x B(32x16 bf16) + C(16x16 f32)
  return __builtin_amdgcn_wmma_f32_16x16x32_bf16(false, a, false, b, (short)0, c,
                                                 false, false);
}

// A-operand fragment of a 16xK row-major tile: lane = row (l16), element j
// covers k = kbyteoff + {8*hl + j%8, 16 + 8*hl + (j%8)} for j<8 / j>=8.
static __device__ __forceinline__ bf16x16 afrag(const __bf16* row, int hl) {
  return bmerge(*(const bf16x8*)(row + 8 * hl),
                *(const bf16x8*)(row + 16 + 8 * hl));
}

// ---------------------------------------------------------------------------
// fp32 -> bf16 conversion (vector-of-8 per thread)
// ---------------------------------------------------------------------------
__global__ void cvt_bf16_kernel(const float* __restrict__ src,
                                __bf16* __restrict__ dst, int n8) {
  int i = blockIdx.x * blockDim.x + threadIdx.x;
  if (i < n8) {
    f32x8 v = *(const f32x8*)(src + (size_t)i * 8);
    bf16x8 o;
#pragma unroll
    for (int r = 0; r < 8; ++r) o[r] = (__bf16)v[r];
    *(bf16x8*)(dst + (size_t)i * 8) = o;
  }
}

// int keep-mask -> additive f32 mask (0 keep, -3e38 drop)
__global__ void mask_kernel(const int* __restrict__ msk,
                            float* __restrict__ mf, int n) {
  int i = blockIdx.x * blockDim.x + threadIdx.x;
  if (i < n) mf[i] = (msk[i] != 0) ? 0.0f : NEGINF;
}

// ---------------------------------------------------------------------------
// Fused QKV projection GEMM: out = hs @ W^T + b   (M=8192, N=1024, K=1024)
// blockIdx.z selects {Q,K,V}. One wave computes a 32(M) x 64(N) tile:
// 8 WMMA per k-step from 12 B128 loads (B fragments reused across 2 M tiles).
// Q is pre-scaled by QSCALE; V is stored transposed [bh][hd][s].
// ---------------------------------------------------------------------------
__global__ __launch_bounds__(256) void qkv_wmma_kernel(
    const __bf16* __restrict__ hsb, const __bf16* __restrict__ wb,
    const float* __restrict__ bq, const float* __restrict__ bk,
    const float* __restrict__ bv, __bf16* __restrict__ Qb,
    __bf16* __restrict__ Kb, __bf16* __restrict__ Vtb) {
  const int lane = threadIdx.x & 31;
  const int wv   = threadIdx.x >> 5;
  const int l16  = lane & 15;
  const int hl   = lane >> 4;      // which 16-lane half
  const int z    = blockIdx.z;     // 0=Q 1=K 2=V

  const int mtile = (blockIdx.x * 8 + wv) * 32;
  const int nbase = blockIdx.y * 64;

  const __bf16* w    = wb + (size_t)z * DMODEL * DMODEL;
  const float*  bias = (z == 0) ? bq : (z == 1) ? bk : bv;

  const __bf16* arow0 = hsb + (size_t)(mtile + l16) * DMODEL;
  const __bf16* arow1 = arow0 + (size_t)16 * DMODEL;

  // B operand: lane holds column n; element j <-> k = k0 + {8*hl + j%8, 16 + 8*hl + j%8} for j<8 / j>=8
  const __bf16* wc[4];
#pragma unroll
  for (int t = 0; t < 4; ++t)
    wc[t] = w + (size_t)(nbase + t * 16 + l16) * DMODEL + 8 * hl;

  f32x8 acc0[4] = {f32x8{}, f32x8{}, f32x8{}, f32x8{}};
  f32x8 acc1[4] = {f32x8{}, f32x8{}, f32x8{}, f32x8{}};

  for (int k0 = 0; k0 < DMODEL; k0 += 32) {
    bf16x16 a0 = afrag(arow0 + k0, hl);
    bf16x16 a1 = afrag(arow1 + k0, hl);
#pragma unroll
    for (int t = 0; t < 4; ++t) {
      bf16x16 bfrag = bmerge(*(const bf16x8*)(wc[t] + k0),
                             *(const bf16x8*)(wc[t] + k0 + 16));
      acc0[t] = wmma_bf16(a0, bfrag, acc0[t]);
      acc1[t] = wmma_bf16(a1, bfrag, acc1[t]);
    }
  }

  const float scale = (z == 0) ? QSCALE : 1.0f;
  const int   bidx  = mtile >> 11;  // mtile / SEQ (32-row tile never splits b)
#pragma unroll
  for (int t = 0; t < 4; ++t) {
    const int   n  = nbase + t * 16 + l16;
    const float bn = bias[n];
    const int   hh = n >> 6, hd = n & (HDIM - 1);
    const int   bh = bidx * NHEADS + hh;
#pragma unroll
    for (int mt = 0; mt < 2; ++mt) {
      const f32x8& a     = mt ? acc1[t] : acc0[t];
      const int    mbase = mtile + mt * 16;
      if (z == 2) {
        // V transposed: [bh][hd][s]; this lane's 8 rows are consecutive s
        const int s0 = (mbase + 8 * hl) & (SEQ - 1);
        bf16x8 pack;
#pragma unroll
        for (int r = 0; r < 8; ++r) pack[r] = (__bf16)(a[r] + bn);
        *(bf16x8*)(Vtb + ((size_t)(bh * HDIM + hd)) * SEQ + s0) = pack;
      } else {
        __bf16* outp = (z == 0) ? Qb : Kb;  // [bh][s][hd]
#pragma unroll
        for (int r = 0; r < 8; ++r) {
          const int srow = (mbase + r + 8 * hl) & (SEQ - 1);
          outp[((size_t)(bh * SEQ + srow)) * HDIM + hd] =
              (__bf16)((a[r] + bn) * scale);
        }
      }
    }
  }
}

// ---------------------------------------------------------------------------
// Online-softmax update for one 16-query column set over a 32-key block.
// Scores are base-2 scaled (Q carries 1/sqrt(HD)*log2e). Returns P^T as the
// WMMA B operand (lane = column q, element j <-> key 16*hl + j).
// ---------------------------------------------------------------------------
static __device__ __forceinline__ bf16x16 softmax_pb(f32x8 st0, f32x8 st1,
                                                     float& mrun, float& lsum,
                                                     f32x8* acc, int hl) {
  float tm = fmaxf(st0[0], st1[0]);
#pragma unroll
  for (int r = 1; r < 8; ++r) tm = fmaxf(tm, fmaxf(st0[r], st1[r]));
  tm = fmaxf(tm, __shfl_xor(tm, 16, 32));
  const float mnew = fmaxf(mrun, tm);
  const float resc = __builtin_amdgcn_exp2f(mrun - mnew);
  mrun = mnew;

  f32x8 p0, p1;
  float ls = 0.0f;
#pragma unroll
  for (int r = 0; r < 8; ++r) {
    p0[r] = __builtin_amdgcn_exp2f(st0[r] - mnew);
    p1[r] = __builtin_amdgcn_exp2f(st1[r] - mnew);
    ls += p0[r] + p1[r];
  }
  ls += __shfl_xor(ls, 16, 32);
  lsum = lsum * resc + ls;
#pragma unroll
  for (int c = 0; c < 4; ++c)
#pragma unroll
    for (int r = 0; r < 8; ++r) acc[c][r] *= resc;

  bf16x16 pb;
#pragma unroll
  for (int r = 0; r < 8; ++r) {
    const float x0 = __shfl_xor(p0[r], 16, 32);
    const float x1 = __shfl_xor(p1[r], 16, 32);
    pb[r]     = (__bf16)(hl ? x1 : p0[r]);
    pb[r + 8] = (__bf16)(hl ? p1[r] : x0);
  }
  return pb;
}

// ---------------------------------------------------------------------------
// Flash attention: one wave owns a 32-query tile of one (b,h); iterates keys
// 32 at a time. Scores computed transposed (S^T = K * Q^T) so softmax
// reductions are per-lane + one shfl_xor(16). Every K and V fragment is
// reused across the two 16-query column sets: 16 WMMA per 32-key block.
// ---------------------------------------------------------------------------
__global__ __launch_bounds__(256) void attn_wmma_kernel(
    const __bf16* __restrict__ Qb, const __bf16* __restrict__ Kb,
    const __bf16* __restrict__ Vtb, const float* __restrict__ maskF,
    float* __restrict__ out) {
  const int lane = threadIdx.x & 31;
  const int wv   = threadIdx.x >> 5;
  const int l16  = lane & 15;
  const int hl   = lane >> 4;

  const int bh    = blockIdx.x >> 3;            // 0..63
  const int qt    = (blockIdx.x & 7) * 8 + wv;  // 0..63
  const int qbase = qt * 32;
  const int bidx  = bh >> 4, hh = bh & 15;

  const __bf16* Qp = Qb + ((size_t)bh * SEQ + qbase) * HDIM;
  const __bf16* Kp = Kb + (size_t)bh * SEQ * HDIM;
  const __bf16* Vp = Vtb + (size_t)bh * HDIM * SEQ;
  const float*  mp = maskF + bidx * SEQ;

  // Q^T as WMMA B operand (fixed): lane = column q; frag f covers hd [32f,+32)
  const __bf16* qrowA = Qp + l16 * HDIM;
  const __bf16* qrowB = qrowA + 16 * HDIM;
  const bf16x16 qA0 = bmerge(*(const bf16x8*)(qrowA + 8 * hl),
                             *(const bf16x8*)(qrowA + 16 + 8 * hl));
  const bf16x16 qA1 = bmerge(*(const bf16x8*)(qrowA + 32 + 8 * hl),
                             *(const bf16x8*)(qrowA + 32 + 16 + 8 * hl));
  const bf16x16 qB0 = bmerge(*(const bf16x8*)(qrowB + 8 * hl),
                             *(const bf16x8*)(qrowB + 16 + 8 * hl));
  const bf16x16 qB1 = bmerge(*(const bf16x8*)(qrowB + 32 + 8 * hl),
                             *(const bf16x8*)(qrowB + 32 + 16 + 8 * hl));

  f32x8 accA[4] = {f32x8{}, f32x8{}, f32x8{}, f32x8{}};  // ctx^T, q 0..15
  f32x8 accB[4] = {f32x8{}, f32x8{}, f32x8{}, f32x8{}};  // ctx^T, q 16..31
  float mrunA = NEGINF, lsumA = 0.0f;
  float mrunB = NEGINF, lsumB = 0.0f;

  for (int kb = 0; kb <= qbase + 31; kb += 32) {
    const __bf16* k0r = Kp + (size_t)(kb + l16) * HDIM;
    const __bf16* k1r = k0r + 16 * HDIM;
    __builtin_prefetch(k0r + 32 * HDIM, 0, 1);  // hint next key block

    // S^T tiles: lane = column q, rows = keys (r + 8*hl [+16])
    f32x8 sA0 = {}, sA1 = {}, sB0 = {}, sB1 = {};
    bf16x16 ka;
    ka  = afrag(k0r, hl);           // keys kb..kb+15, hd 0..31
    sA0 = wmma_bf16(ka, qA0, sA0);
    sB0 = wmma_bf16(ka, qB0, sB0);
    ka  = afrag(k0r + 32, hl);      // keys kb..kb+15, hd 32..63
    sA0 = wmma_bf16(ka, qA1, sA0);
    sB0 = wmma_bf16(ka, qB1, sB0);
    ka  = afrag(k1r, hl);           // keys kb+16..kb+31, hd 0..31
    sA1 = wmma_bf16(ka, qA0, sA1);
    sB1 = wmma_bf16(ka, qB0, sB1);
    ka  = afrag(k1r + 32, hl);      // keys kb+16..kb+31, hd 32..63
    sA1 = wmma_bf16(ka, qA1, sA1);
    sB1 = wmma_bf16(ka, qB1, sB1);

    // additive key-padding mask (base-2 domain is fine: -3e38 -> exp2 = 0)
    const f32x8 ma0 = *(const f32x8*)(mp + kb + 8 * hl);
    const f32x8 ma1 = *(const f32x8*)(mp + kb + 16 + 8 * hl);
#pragma unroll
    for (int r = 0; r < 8; ++r) {
      sA0[r] += ma0[r]; sA1[r] += ma1[r];
      sB0[r] += ma0[r]; sB1[r] += ma1[r];
    }

    if (kb + 31 > qbase) {  // causal boundary for q subtile A (uniform branch)
      const int q = qbase + l16;
#pragma unroll
      for (int r = 0; r < 8; ++r) {
        const int key0 = kb + r + 8 * hl;
        if (key0 > q)      sA0[r] = NEGINF;
        if (key0 + 16 > q) sA1[r] = NEGINF;
      }
    }
    if (kb + 31 > qbase + 16) {  // causal boundary for q subtile B
      const int q = qbase + 16 + l16;
#pragma unroll
      for (int r = 0; r < 8; ++r) {
        const int key0 = kb + r + 8 * hl;
        if (key0 > q)      sB0[r] = NEGINF;
        if (key0 + 16 > q) sB1[r] = NEGINF;
      }
    }

    const bf16x16 pbA = softmax_pb(sA0, sA1, mrunA, lsumA, accA, hl);
    const bf16x16 pbB = softmax_pb(sB0, sB1, mrunB, lsumB, accB, hl);

    // ctx^T += V^T(16hd x 32key) * P^T(32key x 16q); V frags reused A/B
#pragma unroll
    for (int c = 0; c < 4; ++c) {
      const __bf16* vrow = Vp + (size_t)(c * 16 + l16) * SEQ + kb;
      bf16x16 va = bmerge(*(const bf16x8*)(vrow + 16 * hl),
                          *(const bf16x8*)(vrow + 16 * hl + 8));
      accA[c] = wmma_bf16(va, pbA, accA[c]);
      accB[c] = wmma_bf16(va, pbB, accB[c]);
    }
  }

  // normalize + store: out[b][s=q][h*64+hd], lane's 8 hd values contiguous
  const float invA = 1.0f / lsumA;
  const float invB = 1.0f / lsumB;
  float* orowA =
      out + ((size_t)(bidx * SEQ + qbase + l16)) * DMODEL + hh * HDIM;
  float* orowB = orowA + (size_t)16 * DMODEL;
#pragma unroll
  for (int c = 0; c < 4; ++c) {
    f32x8 oA, oB;
#pragma unroll
    for (int r = 0; r < 8; ++r) {
      oA[r] = accA[c][r] * invA;
      oB[r] = accB[c][r] * invB;
    }
    *(f32x8*)(orowA + c * 16 + 8 * hl) = oA;
    *(f32x8*)(orowB + c * 16 + 8 * hl) = oB;
  }
}

// ---------------------------------------------------------------------------
extern "C" void kernel_launch(void* const* d_in, const int* in_sizes, int n_in,
                              void* d_out, int out_size, void* d_ws,
                              size_t ws_size, hipStream_t stream) {
  const float* hs  = (const float*)d_in[0];
  const int*   msk = (const int*)d_in[1];
  const float* Wq  = (const float*)d_in[2];
  const float* bq  = (const float*)d_in[3];
  const float* Wk  = (const float*)d_in[4];
  const float* bk  = (const float*)d_in[5];
  const float* Wv  = (const float*)d_in[6];
  const float* bv  = (const float*)d_in[7];
  float* out = (float*)d_out;

  const size_t HS_E = (size_t)MROWS * DMODEL;   // 8,388,608
  const size_t W_E  = (size_t)DMODEL * DMODEL;  // 1,048,576

  // Workspace layout (bf16 regions then f32 mask): ~73.5 MB total
  __bf16* hsb   = (__bf16*)d_ws;
  __bf16* wb    = hsb + HS_E;       // 3 weight matrices, concat
  __bf16* Qb    = wb + 3 * W_E;     // [bh][s][hd]
  __bf16* Kb    = Qb + HS_E;        // [bh][s][hd]
  __bf16* Vtb   = Kb + HS_E;        // [bh][hd][s]
  float*  maskF = (float*)(Vtb + HS_E);

  cvt_bf16_kernel<<<dim3((unsigned)((HS_E / 8 + 255) / 256)), 256, 0, stream>>>(
      hs, hsb, (int)(HS_E / 8));
  cvt_bf16_kernel<<<dim3((unsigned)((W_E / 8 + 255) / 256)), 256, 0, stream>>>(
      Wq, wb + 0 * W_E, (int)(W_E / 8));
  cvt_bf16_kernel<<<dim3((unsigned)((W_E / 8 + 255) / 256)), 256, 0, stream>>>(
      Wk, wb + 1 * W_E, (int)(W_E / 8));
  cvt_bf16_kernel<<<dim3((unsigned)((W_E / 8 + 255) / 256)), 256, 0, stream>>>(
      Wv, wb + 2 * W_E, (int)(W_E / 8));
  mask_kernel<<<dim3((BATCH * SEQ + 255) / 256), 256, 0, stream>>>(
      msk, maskF, BATCH * SEQ);

  // QKV: grid = (M tiles / 8 waves, N/64, 3 matrices), 256 threads = 8 waves
  qkv_wmma_kernel<<<dim3(MROWS / (32 * 8), DMODEL / 64, 3), 256, 0, stream>>>(
      hsb, wb, bq, bk, bv, Qb, Kb, Vtb);

  // Attention: 64 (b,h) * 8 chunks of 8 q-tiles (32 queries each)
  attn_wmma_kernel<<<dim3((BATCH * NHEADS) * (SEQ / 32 / 8)), 256, 0, stream>>>(
      Qb, Kb, Vtb, maskF, out);
}
